// SparseAttention_38027640439432
// MI455X (gfx1250) — matrix operand order repaired and checked by reading, not compiled, not run on hardware
//
#include <hip/hip_runtime.h>
#include <hip/hip_bf16.h>

// ---------------------------------------------------------------------------
// Sliding-window attention on MI455X (gfx1250, wave32, WMMA).
//   - All GEMM work via V_WMMA_F32_16X16X32_BF16 (fp32 accumulate): 16 KFLOP
//     per issue vs 2 KFLOP for the f32 K=4 WMMA -> right pick for a
//     compute-bound 155 GFLOP problem whose working set fits in 192 MB L2.
//   - RoPE cos/sin precomputed once (256K unique values) instead of per-element
//     transcendentals in the GEMM epilogue.
//   - Attention K/V windows staged global->LDS with GLOBAL_LOAD_ASYNC_TO_LDS
//     (ASYNCcnt + s_wait_asynccnt), fragments then read from LDS.
// Workspace: qb|kb|vt|ao bf16 (32 MB each) + cos/sin tables (2 MB) = 130 MB.
// ---------------------------------------------------------------------------

#define DMODEL 1024
#define NHEADS 16
#define DHEAD  64
#define LSEQ   8192
#define BATCH  2
#define MTOT   (BATCH * LSEQ)   // 16384 rows

// LDS pitches (elements) - padded to dodge bank conflicts, 16B-aligned rows.
#define KLP 72    // K window  [256][64]  bf16, pitch 144 B
#define VLP 264   // V window  [64][256]  bf16, pitch 528 B
#define SPP 264   // scores    [64][256]  f32 / bf16

typedef __attribute__((ext_vector_type(16))) __bf16 v16bf;
typedef __attribute__((ext_vector_type(8)))  float  v8f;

static __device__ __forceinline__ v8f wmma_bf16(v16bf a, v16bf b, v8f c) {
    // (neg_a, A, neg_b, B, c_mod, C, reuse_a, reuse_b)
    return __builtin_amdgcn_wmma_f32_16x16x32_bf16(false, a, false, b,
                                                   (short)0, c, false, false);
}

// ---- CDNA5 async global->LDS copy (CDNA5 ISA) --------
// GVS mode: addr = SGPR base + 32-bit VGPR offset. VDST vgpr = LDS byte addr.
// 16 bytes per lane per issue; tracked with ASYNCcnt.
static __device__ __forceinline__ void async_g2l_b128(unsigned lds_byte,
                                                      const void* sbase,
                                                      int goffset_bytes) {
    asm volatile("global_load_async_to_lds_b128 %0, %1, %2"
                 :: "v"(lds_byte), "v"(goffset_bytes), "s"(sbase)
                 : "memory");
}
static __device__ __forceinline__ void wait_async0() {
    asm volatile("s_wait_asynccnt 0x0" ::: "memory");
}
// Generic LDS pointer -> LDS byte offset (aperture puts offset in addr[31:0]).
static __device__ __forceinline__ unsigned lds_off(const void* p) {
    return (unsigned)(unsigned long long)p;
}

// ---- fragment loaders (CDNA5 16-bit layouts, CDNA5 ISA §7.12.2) --
// A 16x32: lanes 0-15 = rows, elems 0..7 -> K 0..7, 8..15 -> K 16..23;
//          lanes 16-31 same rows, K 8..15 / 24..31.
static __device__ __forceinline__ v16bf frag_a_f32(const float* p, int ld, int lane) {
    int m = lane & 15, sel = lane >> 4;
    const float* r0 = p + (size_t)m * ld + sel * 8;
    const float* r1 = p + (size_t)m * ld + 16 + sel * 8;
    v16bf r;
#pragma unroll
    for (int i = 0; i < 8; ++i) { r[i] = (__bf16)r0[i]; r[8 + i] = (__bf16)r1[i]; }
    return r;
}
static __device__ __forceinline__ v16bf frag_a_bf16(const __bf16* p, int ld, int lane) {
    int m = lane & 15, sel = lane >> 4;
    const __bf16* r0 = p + (size_t)m * ld + sel * 8;
    const __bf16* r1 = p + (size_t)m * ld + 16 + sel * 8;
    v16bf r;
#pragma unroll
    for (int i = 0; i < 8; ++i) { r[i] = r0[i]; r[8 + i] = r1[i]; }
    return r;
}
// B 32x16 (KxN): lane n<16 holds K 0..7 / 16..23 of column n; lane n+16 holds K 8..15 / 24..31.
static __device__ __forceinline__ v16bf frag_b_f32(const float* p, int ld, int lane) {
    int n = lane & 15, sel = lane >> 4;
    const float* r0 = p + (size_t)n * ld + sel * 8;
    const float* r1 = p + (size_t)n * ld + 16 + sel * 8;
    v16bf r;
#pragma unroll
    for (int i = 0; i < 8; ++i) { r[i] = (__bf16)r0[i]; r[8 + i] = (__bf16)r1[i]; }
    return r;
}
static __device__ __forceinline__ v16bf frag_b_bf16(const __bf16* p, int ld, int lane) {
    int n = lane & 15, sel = lane >> 4;
    const __bf16* r0 = p + (size_t)n * ld + sel * 8;
    const __bf16* r1 = p + (size_t)n * ld + 16 + sel * 8;
    v16bf r;
#pragma unroll
    for (int i = 0; i < 8; ++i) { r[i] = r0[i]; r[8 + i] = r1[i]; }
    return r;
}
// C/D 16x16 f32: VGPR j, lanes 0-15 -> (M=j, N=lane); lanes 16-31 -> (M=8+j).
static __device__ __forceinline__ void store_acc(float* base, int pitch, v8f c, int lane) {
    int n = lane & 15, mo = (lane >> 4) << 3;
#pragma unroll
    for (int j = 0; j < 8; ++j) base[(size_t)(mo + j) * pitch + n] = c[j];
}

// ---------------------------------------------------------------------------
// Kernel 0: RoPE tables cos/sin[l][d'] for l<8192, d'<32 (the only 256K unique
// angle values) - keeps transcendentals out of the GEMM epilogue.
// ---------------------------------------------------------------------------
__global__ __launch_bounds__(256) void rope_table_kernel(float* __restrict__ ct,
                                                         float* __restrict__ st) {
    int i = blockIdx.x * 256 + threadIdx.x;     // i < LSEQ*32
    int l = i >> 5, d = i & 31;
    float ang = (float)l * powf(10000.0f, -(float)(2 * d) * (1.0f / 64.0f));
    float s, c;
    sincosf(ang, &s, &c);
    ct[i] = c;
    st[i] = s;
}

// ---------------------------------------------------------------------------
// Kernel 1: fused QKV projection + RoPE.
// grid = (MTOT/128, 48): y covers 3 projections x 16 heads (tile == one head,
// so the rotate_half pairing d <-> d +/- 32 stays inside the block's tile).
// ---------------------------------------------------------------------------
__global__ __launch_bounds__(256) void qkv_rope_kernel(
    const float* __restrict__ x, const float* __restrict__ wq,
    const float* __restrict__ wk, const float* __restrict__ wv,
    const float* __restrict__ ct, const float* __restrict__ st,
    __bf16* __restrict__ qb, __bf16* __restrict__ kb, __bf16* __restrict__ vt) {
    __shared__ float sm[128][68];
    const int tid = threadIdx.x, lane = tid & 31, wid = tid >> 5;
    const int m0 = blockIdx.x * 128;
    const int proj = blockIdx.y >> 4;   // 0=q 1=k 2=v
    const int h    = blockIdx.y & 15;
    const float* w = (proj == 0) ? wq : (proj == 1) ? wk : wv;
    const int nw0 = h * DHEAD;
    const int wm = wid & 3, wn = wid >> 2;   // 4x2 wave grid -> 128x64 tile

    v8f acc[2][2] = {};
    for (int k0 = 0; k0 < DMODEL; k0 += 32) {
        v16bf a0 = frag_a_f32(x + (size_t)(m0 + wm * 32)      * DMODEL + k0, DMODEL, lane);
        v16bf a1 = frag_a_f32(x + (size_t)(m0 + wm * 32 + 16) * DMODEL + k0, DMODEL, lane);
        v16bf b0 = frag_b_f32(w + (size_t)(nw0 + wn * 32)      * DMODEL + k0, DMODEL, lane);
        v16bf b1 = frag_b_f32(w + (size_t)(nw0 + wn * 32 + 16) * DMODEL + k0, DMODEL, lane);
        acc[0][0] = wmma_bf16(a0, b0, acc[0][0]);
        acc[0][1] = wmma_bf16(a0, b1, acc[0][1]);
        acc[1][0] = wmma_bf16(a1, b0, acc[1][0]);
        acc[1][1] = wmma_bf16(a1, b1, acc[1][1]);
    }
#pragma unroll
    for (int mt = 0; mt < 2; ++mt)
#pragma unroll
        for (int nt = 0; nt < 2; ++nt)
            store_acc(&sm[wm * 32 + mt * 16][wn * 32 + nt * 16], 68, acc[mt][nt], lane);
    __syncthreads();

    for (int i = tid; i < 128 * 64; i += 256) {
        int r = i >> 6, d = i & 63;
        int m = m0 + r, b = m >> 13, l = m & (LSEQ - 1);
        float y = sm[r][d];
        if (proj < 2) {
            float cv = ct[l * 32 + (d & 31)];
            float sv = st[l * 32 + (d & 31)];
            float rot = (d < 32) ? -sm[r][d + 32] : sm[r][d - 32];
            y = y * cv + rot * sv;
        }
        size_t bh = (size_t)(b * NHEADS + h);
        if (proj == 0)      qb[(bh * LSEQ + l) * DHEAD + d] = (__bf16)y;
        else if (proj == 1) kb[(bh * LSEQ + l) * DHEAD + d] = (__bf16)y;
        else                vt[(bh * DHEAD + d) * LSEQ + l] = (__bf16)y;  // transposed
    }
}

// ---------------------------------------------------------------------------
// Kernel 2: windowed attention. Block = 64 queries x 256-key window.
// K/V windows staged via async global->LDS b128 copies; edge clamping is done
// per 16-byte chunk (the +/-64-key invalid margins are chunk-aligned, and
// invalid keys are masked to -1e30 / get P==0, so clamped data is inert).
// grid = B*H*128.
// ---------------------------------------------------------------------------
__global__ __launch_bounds__(256) void attn_kernel(
    const __bf16* __restrict__ qb, const __bf16* __restrict__ kb,
    const __bf16* __restrict__ vt, __bf16* __restrict__ ao) {
    extern __shared__ char smraw[];
    __bf16* Kl = (__bf16*)smraw;                                  // [256][KLP]
    __bf16* Vl = (__bf16*)(smraw + 256 * KLP * 2);                // [64][VLP]
    float*  S  = (float*)(smraw + 256 * KLP * 2 + 64 * VLP * 2);  // [64][SPP]
    __bf16* P  = (__bf16*)((char*)S + (size_t)64 * SPP * 4);      // [64][SPP]

    const int tid = threadIdx.x, lane = tid & 31, wid = tid >> 5;
    int gid = blockIdx.x;
    int b = gid >> 11, rem = gid & 2047;
    int h = rem >> 7,  t   = rem & 127;
    int win = t >> 1,  hs  = t & 1;
    int l0 = win * 128 + hs * 64;
    int kstart = win * 128 - 64;
    size_t bh = (size_t)(b * NHEADS + h);
    const __bf16* qbase = qb + (bh * LSEQ + l0) * DHEAD;
    const __bf16* kgbase = kb + bh * LSEQ * DHEAD;
    const __bf16* vgbase = vt + bh * DHEAD * LSEQ;

    // ---- async stage K window [256 keys][64] and V window [64][256 keys] ----
    {
        unsigned k0l = lds_off(Kl), v0l = lds_off(Vl);
        for (int c = tid; c < 2048; c += 256) {       // K: 256 rows x 8 chunks
            int row = c >> 3, ch = c & 7;
            int gkey = kstart + row;
            gkey = gkey < 0 ? 0 : (gkey > LSEQ - 1 ? LSEQ - 1 : gkey);
            async_g2l_b128(k0l + (unsigned)(row * (KLP * 2) + ch * 16),
                           kgbase, (gkey * DHEAD + ch * 8) * 2);
        }
        for (int c = tid; c < 2048; c += 256) {       // V: 64 rows x 32 chunks
            int d = c >> 5, ch = c & 31;
            int gk = kstart + ch * 8;
            gk = gk < 0 ? 0 : (gk > LSEQ - 8 ? LSEQ - 8 : gk);
            async_g2l_b128(v0l + (unsigned)(d * (VLP * 2) + ch * 16),
                           vgbase, (d * LSEQ + gk) * 2);
        }
        wait_async0();       // own wave's copies done ...
        __syncthreads();     // ... and every wave has waited -> tile complete
    }

    // ---- S = (q . k^T) / 8, masked ----
    {
        int wm = wid & 1, wn = wid >> 1;   // 2x4 wave grid over 64x256
        v8f acc[2][4] = {};
        for (int k0 = 0; k0 < DHEAD; k0 += 32) {
            v16bf a0 = frag_a_bf16(qbase + (size_t)(wm * 32)      * DHEAD + k0, DHEAD, lane);
            v16bf a1 = frag_a_bf16(qbase + (size_t)(wm * 32 + 16) * DHEAD + k0, DHEAD, lane);
#pragma unroll
            for (int nt = 0; nt < 4; ++nt) {
                v16bf bf = frag_b_bf16(Kl + (size_t)(wn * 64 + nt * 16) * KLP + k0, KLP, lane);
                acc[0][nt] = wmma_bf16(a0, bf, acc[0][nt]);
                acc[1][nt] = wmma_bf16(a1, bf, acc[1][nt]);
            }
        }
        int n = lane & 15, mo = (lane >> 4) << 3;
#pragma unroll
        for (int mt = 0; mt < 2; ++mt)
#pragma unroll
            for (int nt = 0; nt < 4; ++nt)
#pragma unroll
                for (int j = 0; j < 8; ++j) {
                    int row = wm * 32 + mt * 16 + mo + j;
                    int col = wn * 64 + nt * 16 + n;
                    int kk = kstart + col;
                    float v = (kk >= 0 && kk < LSEQ) ? acc[mt][nt][j] * 0.125f : -1e30f;
                    S[(size_t)row * SPP + col] = v;
                }
    }
    __syncthreads();

    // ---- softmax rows (thread r handles row r) ----
    if (tid < 64) {
        float mx = -1e30f;
        for (int c = 0; c < 256; ++c) mx = fmaxf(mx, S[(size_t)tid * SPP + c]);
        float sum = 0.0f;
        for (int c = 0; c < 256; ++c) {
            float e = __expf(S[(size_t)tid * SPP + c] - mx);
            S[(size_t)tid * SPP + c] = e;
            sum += e;
        }
        float inv = 1.0f / sum;
        for (int c = 0; c < 256; ++c)
            P[(size_t)tid * SPP + c] = (__bf16)(S[(size_t)tid * SPP + c] * inv);
    }
    __syncthreads();

    // ---- O = P . V  (V staged transposed: B-frag columns contiguous in LDS) --
    {
        int wm = wid & 3, wn = wid >> 2;   // 4x2 wave grid over 64x64
        v8f acc[2] = {};
        for (int k0 = 0; k0 < 256; k0 += 32) {
            v16bf af = frag_a_bf16(P + (size_t)(wm * 16) * SPP + k0, SPP, lane);
#pragma unroll
            for (int nt = 0; nt < 2; ++nt) {
                v16bf bf = frag_b_bf16(Vl + (size_t)(wn * 32 + nt * 16) * VLP + k0, VLP, lane);
                acc[nt] = wmma_bf16(af, bf, acc[nt]);
            }
        }
        int n = lane & 15, mo = (lane >> 4) << 3;
#pragma unroll
        for (int nt = 0; nt < 2; ++nt)
#pragma unroll
            for (int j = 0; j < 8; ++j) {
                int l = l0 + wm * 16 + mo + j;
                int d = wn * 32 + nt * 16 + n;
                ao[((size_t)b * LSEQ + l) * DMODEL + h * DHEAD + d] = (__bf16)acc[nt][j];
            }
    }
}

// ---------------------------------------------------------------------------
// Kernel 3: output projection Y = ao @ wo^T + bo (fp32 out).
// grid = (MTOT/128, DMODEL/64)
// ---------------------------------------------------------------------------
__global__ __launch_bounds__(256) void outproj_kernel(
    const __bf16* __restrict__ ao, const float* __restrict__ wo,
    const float* __restrict__ bo, float* __restrict__ out) {
    const int tid = threadIdx.x, lane = tid & 31, wid = tid >> 5;
    const int m0 = blockIdx.x * 128, n0 = blockIdx.y * 64;
    int wm = wid & 3, wn = wid >> 2;
    v8f acc[2][2] = {};
    for (int k0 = 0; k0 < DMODEL; k0 += 32) {
        v16bf a0 = frag_a_bf16(ao + (size_t)(m0 + wm * 32)      * DMODEL + k0, DMODEL, lane);
        v16bf a1 = frag_a_bf16(ao + (size_t)(m0 + wm * 32 + 16) * DMODEL + k0, DMODEL, lane);
        v16bf b0 = frag_b_f32(wo + (size_t)(n0 + wn * 32)      * DMODEL + k0, DMODEL, lane);
        v16bf b1 = frag_b_f32(wo + (size_t)(n0 + wn * 32 + 16) * DMODEL + k0, DMODEL, lane);
        acc[0][0] = wmma_bf16(a0, b0, acc[0][0]);
        acc[0][1] = wmma_bf16(a0, b1, acc[0][1]);
        acc[1][0] = wmma_bf16(a1, b0, acc[1][0]);
        acc[1][1] = wmma_bf16(a1, b1, acc[1][1]);
    }
    int n = lane & 15, mo = (lane >> 4) << 3;
#pragma unroll
    for (int mt = 0; mt < 2; ++mt)
#pragma unroll
        for (int nt = 0; nt < 2; ++nt)
#pragma unroll
            for (int j = 0; j < 8; ++j) {
                int row = m0 + wm * 32 + mt * 16 + mo + j;
                int col = n0 + wn * 32 + nt * 16 + n;
                out[(size_t)row * DMODEL + col] = acc[mt][nt][j] + bo[col];
            }
}

// ---------------------------------------------------------------------------
extern "C" void kernel_launch(void* const* d_in, const int* in_sizes, int n_in,
                              void* d_out, int out_size, void* d_ws, size_t ws_size,
                              hipStream_t stream) {
    const float* x  = (const float*)d_in[0];
    const float* wq = (const float*)d_in[1];
    const float* wk = (const float*)d_in[2];
    const float* wv = (const float*)d_in[3];
    const float* wo = (const float*)d_in[4];
    const float* bo = (const float*)d_in[5];
    float* out = (float*)d_out;

    // workspace: 4 x 32 MB bf16 + 2 x 1 MB f32 RoPE tables (needs >= 130 MB)
    __bf16* qb = (__bf16*)d_ws;
    __bf16* kb = qb + (size_t)MTOT * DMODEL;
    __bf16* vt = kb + (size_t)MTOT * DMODEL;
    __bf16* ao = vt + (size_t)MTOT * DMODEL;
    float*  ct = (float*)(ao + (size_t)MTOT * DMODEL);
    float*  st = ct + (size_t)LSEQ * 32;

    rope_table_kernel<<<dim3(LSEQ * 32 / 256), 256, 0, stream>>>(ct, st);

    qkv_rope_kernel<<<dim3(MTOT / 128, 48), 256, 0, stream>>>(x, wq, wk, wv,
                                                              ct, st, qb, kb, vt);

    size_t smem = (size_t)256 * KLP * 2 + (size_t)64 * VLP * 2
                + (size_t)64 * SPP * 4 + (size_t)64 * SPP * 2;   // ~168 KB of 320 KB
    attn_kernel<<<dim3(BATCH * NHEADS * 128), 256, smem, stream>>>(qb, kb, vt, ao);

    outproj_kernel<<<dim3(MTOT / 128, DMODEL / 64), 256, 0, stream>>>(ao, wo, bo, out);
}
